// AnomalyAttention_62070867361927
// MI455X (gfx1250) — matrix operand order repaired and checked by reading, not compiled, not run on hardware
//
#include <hip/hip_runtime.h>

// AnomalyAttention for MI455X (gfx1250, wave32, WMMA f32_16x16x32_f16)
// B=2, L=2048, H=8, E=64

#define BB 2
#define LL 2048
#define HH 8
#define EE 64

typedef _Float16 v2h  __attribute__((ext_vector_type(2)));
typedef _Float16 v8h  __attribute__((ext_vector_type(8)));
typedef _Float16 v16h __attribute__((ext_vector_type(16)));
typedef float    v4f  __attribute__((ext_vector_type(4)));
typedef float    v8f  __attribute__((ext_vector_type(8)));

static __device__ __forceinline__ v8f wmma_f16(v16h a, v16h b, v8f c) {
  // D = A(16x32 f16) x B(32x16 f16) + C(16x16 f32)
  return __builtin_amdgcn_wmma_f32_16x16x32_f16(false, a, false, b, (short)0, c,
                                                false, false);
}

static __device__ __forceinline__ v16h cat8(v8h lo, v8h hi) {
  return __builtin_shufflevector(lo, hi, 0, 1, 2, 3, 4, 5, 6, 7,
                                 8, 9, 10, 11, 12, 13, 14, 15);
}

static __device__ __forceinline__ v8f zero8() {
  v8f z = {0.f, 0.f, 0.f, 0.f, 0.f, 0.f, 0.f, 0.f};
  return z;
}

// Cross-half (lane ^ 16) exchange: pure-VALU v_permlanex16_b32 instead of
// ds_bpermute (avoids DScnt waits on the softmax critical path).
static __device__ __forceinline__ float xhalf(float x) {
#if __has_builtin(__builtin_amdgcn_permlanex16)
  int xi = __float_as_int(x);
  int r  = __builtin_amdgcn_permlanex16(xi, xi, 0x76543210, 0xfedcba98,
                                        false, false);
  return __int_as_float(r);
#else
  return __shfl_xor(x, 16);
#endif
}

// Stage one 32-key x 64-e K tile (f32 global -> f16 LDS, [s][e] layout).
// 128 threads: thread t handles key row s=t>>2, 16 e's at e0=(t&3)*16.
static __device__ __forceinline__ void stage_k_tile(const float* __restrict__ Kb,
                                                    int s0, _Float16* sK, int t) {
  const int s  = t >> 2;
  const int e0 = (t & 3) * 16;
  const float* src = Kb + (size_t)(s0 + s) * (HH * EE) + e0;
  v4f f0 = ((const v4f*)src)[0];
  v4f f1 = ((const v4f*)src)[1];
  v4f f2 = ((const v4f*)src)[2];
  v4f f3 = ((const v4f*)src)[3];
  if (s0 + 32 < LL) __builtin_prefetch((const void*)(src + 32 * HH * EE), 0, 3);
  v8h h0 = {(_Float16)f0.x, (_Float16)f0.y, (_Float16)f0.z, (_Float16)f0.w,
            (_Float16)f1.x, (_Float16)f1.y, (_Float16)f1.z, (_Float16)f1.w};
  v8h h1 = {(_Float16)f2.x, (_Float16)f2.y, (_Float16)f2.z, (_Float16)f2.w,
            (_Float16)f3.x, (_Float16)f3.y, (_Float16)f3.z, (_Float16)f3.w};
  *(v8h*)&sK[s * 64 + e0]     = h0;
  *(v8h*)&sK[s * 64 + e0 + 8] = h1;
}

// Stage one 32-key x 64-e V tile transposed (f16 LDS, [e][s] layout, row=32 halves).
static __device__ __forceinline__ void stage_v_tile(const float* __restrict__ Vb,
                                                    int s0, _Float16* sVT, int t) {
  const int e  = t & 63;
  const int hf = t >> 6;  // 0..1
  if (s0 + 32 < LL)
    __builtin_prefetch((const void*)(Vb + (size_t)(s0 + 32 + hf * 16) * (HH * EE) + e), 0, 3);
#pragma unroll
  for (int p = 0; p < 8; ++p) {
    const int pair = hf * 8 + p;     // 0..15
    const int srel = 2 * pair;       // 0..30
    const float* src = Vb + (size_t)(s0 + srel) * (HH * EE) + e;
    float a = src[0];
    float c = src[HH * EE];
    v2h pk = {(_Float16)a, (_Float16)c};
    *(v2h*)&sVT[e * 32 + srel] = pk;
  }
}

// S^T tiles for 32 keys: c0 covers keys [s0, s0+16), c1 covers [s0+16, s0+32).
// C layout: lane holds query m=(lane&15); element v -> key srel = v + 8*(lane>>4) (+16 for c1).
static __device__ __forceinline__ void qk_tiles(const _Float16* sK, int m, int hiL,
                                                v16h bq0, v16h bq1, v8f& c0, v8f& c1) {
  const _Float16* r0p = sK + (m)      * 64;
  const _Float16* r1p = sK + (16 + m) * 64;
  v16h a;
  a  = cat8(*(const v8h*)(r0p + 0  + 8 * hiL), *(const v8h*)(r0p + 16 + 8 * hiL));
  c0 = wmma_f16(a, bq0, c0);
  a  = cat8(*(const v8h*)(r0p + 32 + 8 * hiL), *(const v8h*)(r0p + 48 + 8 * hiL));
  c0 = wmma_f16(a, bq1, c0);
  a  = cat8(*(const v8h*)(r1p + 0  + 8 * hiL), *(const v8h*)(r1p + 16 + 8 * hiL));
  c1 = wmma_f16(a, bq0, c1);
  a  = cat8(*(const v8h*)(r1p + 32 + 8 * hiL), *(const v8h*)(r1p + 48 + 8 * hiL));
  c1 = wmma_f16(a, bq1, c1);
}

__global__ __launch_bounds__(128)
void anomaly_attn_kernel(const float* __restrict__ Q, const float* __restrict__ K,
                         const float* __restrict__ V, float* __restrict__ outV,
                         float* __restrict__ series) {
  // Double-buffered staging: one barrier per tile, stage(ct+1) overlaps compute(ct).
  __shared__ __align__(16) _Float16 sK[2][32 * 64];
  __shared__ __align__(16) _Float16 sVT[2][64 * 32];

  const int rblk = blockIdx.x & 31;   // 32 row-blocks of 64 rows
  const int bh   = blockIdx.x >> 5;   // 0..15
  const int b = bh >> 3, h = bh & 7;
  const int t    = threadIdx.x;
  const int lane = t & 31;
  const int wv   = t >> 5;            // 4 waves, 16 query rows each
  const int m    = lane & 15;
  const int hiL  = lane >> 4;
  const int r0w  = rblk * 64 + wv * 16;

  const float* Kb = K + (size_t)b * LL * HH * EE + (size_t)h * EE;
  const float* Vb = V + (size_t)b * LL * HH * EE + (size_t)h * EE;

  // Q^T B-operand registers (scale 1/sqrt(64) folded in).
  // B layout: lane = query column n; element j -> k(=e) = ko + 8*hiL + j (j<8), ko + 16 + 8*hiL + (j-8) (j>=8).
  v16h bq0, bq1;
  {
    const float* qrow = Q + ((size_t)(b * LL + r0w + m) * HH + h) * EE;
    float t0[16], t1[16];
#pragma unroll
    for (int c = 0; c < 4; ++c) {
      ((v4f*)t0)[c] = ((const v4f*)(qrow + 8 * hiL + 8 * (c >> 1)))[c];
      ((v4f*)t1)[c] = ((const v4f*)(qrow + 32 + 8 * hiL + 8 * (c >> 1)))[c];
    }
#pragma unroll
    for (int j = 0; j < 16; ++j) {
      bq0[j] = (_Float16)(t0[j] * 0.125f);
      bq1[j] = (_Float16)(t1[j] * 0.125f);
    }
  }

  // ---- Pass 1: online row max + sum-of-exp (scalars only, nothing stored) ----
  float rm = -1e30f, lsum = 0.0f;
  stage_k_tile(Kb, 0, sK[0], t);
  for (int ct = 0; ct < 64; ++ct) {
    __syncthreads();  // stage of tile ct (opposite buffer writes) now visible
    if (ct + 1 < 64) stage_k_tile(Kb, (ct + 1) * 32, sK[(ct + 1) & 1], t);

    v8f c0 = zero8(), c1 = zero8();
    qk_tiles(sK[ct & 1], m, hiL, bq0, bq1, c0, c1);

    float tmax = -1e30f;
#pragma unroll
    for (int v = 0; v < 8; ++v) {
      tmax = fmaxf(tmax, c0[v]);
      tmax = fmaxf(tmax, c1[v]);
    }
    tmax = fmaxf(tmax, xhalf(tmax));
    float nm   = fmaxf(rm, tmax);
    float corr = __expf(rm - nm);
    float ts   = 0.f;
#pragma unroll
    for (int v = 0; v < 8; ++v) {
      ts += __expf(c0[v] - nm);
      ts += __expf(c1[v] - nm);
    }
    ts += xhalf(ts);
    lsum = lsum * corr + ts;
    rm   = nm;
  }
  const float invl = 1.0f / lsum;

  // ---- Pass 2: recompute S, emit normalized series (streaming), and P@V ----
  v8f acc[4];
#pragma unroll
  for (int q = 0; q < 4; ++q) acc[q] = zero8();

  float* srow = series + ((size_t)bh * LL + (r0w + m)) * LL;

  stage_k_tile(Kb, 0, sK[0], t);
  stage_v_tile(Vb, 0, sVT[0], t);
  for (int ct = 0; ct < 64; ++ct) {
    __syncthreads();
    if (ct + 1 < 64) {
      stage_k_tile(Kb, (ct + 1) * 32, sK[(ct + 1) & 1], t);
      stage_v_tile(Vb, (ct + 1) * 32, sVT[(ct + 1) & 1], t);
    }

    v8f c0 = zero8(), c1 = zero8();
    qk_tiles(sK[ct & 1], m, hiL, bq0, bq1, c0, c1);

    float p0a[8], p1a[8];
#pragma unroll
    for (int v = 0; v < 8; ++v) {
      p0a[v] = __expf(c0[v] - rm) * invl;
      p1a[v] = __expf(c1[v] - rm) * invl;
    }

    // series store: lane's row is fixed, keys contiguous -> 4x16B NT stores
    const int s0 = ct * 32;
    v4f f;
    f = (v4f){p0a[0], p0a[1], p0a[2], p0a[3]};
    __builtin_nontemporal_store(f, (v4f*)(srow + s0 + 8 * hiL));
    f = (v4f){p0a[4], p0a[5], p0a[6], p0a[7]};
    __builtin_nontemporal_store(f, (v4f*)(srow + s0 + 8 * hiL + 4));
    f = (v4f){p1a[0], p1a[1], p1a[2], p1a[3]};
    __builtin_nontemporal_store(f, (v4f*)(srow + s0 + 16 + 8 * hiL));
    f = (v4f){p1a[4], p1a[5], p1a[6], p1a[7]};
    __builtin_nontemporal_store(f, (v4f*)(srow + s0 + 16 + 8 * hiL + 4));

    // P is already in A-operand layout per-lane: elems 0..7 <- c0, 8..15 <- c1
    v16h pa;
#pragma unroll
    for (int j = 0; j < 8; ++j) {
      pa[j]     = (_Float16)p0a[j];
      pa[j + 8] = (_Float16)p1a[j];
    }

    // P(16x32) @ V(32x16 per e-chunk), B from transposed-V LDS
#pragma unroll
    for (int q = 0; q < 4; ++q) {
      const _Float16* vp = sVT[ct & 1] + (q * 16 + m) * 32 + 8 * hiL;
      v16h bv = cat8(*(const v8h*)vp, *(const v8h*)(vp + 16));
      acc[q] = wmma_f16(pa, bv, acc[q]);
    }
  }

  // ---- Epilogue: V_out, C layout row = r0w + v + 8*hiL, col e = q*16 + m ----
#pragma unroll
  for (int q = 0; q < 4; ++q) {
#pragma unroll
    for (int v = 0; v < 8; ++v) {
      const int row = r0w + v + 8 * hiL;
      outV[((size_t)(b * LL + row) * HH + h) * EE + q * 16 + m] = acc[q][v];
    }
  }
}

__global__ __launch_bounds__(256)
void anomaly_prior_kernel(const float* __restrict__ sigma, float* __restrict__ prior,
                          float* __restrict__ sigb) {
  const int l  = blockIdx.x & (LL - 1);
  const int bh = blockIdx.x >> 11;
  const int b = bh >> 3, h = bh & 7;

  const float sraw = sigma[((size_t)(b * LL + l)) * HH + h];
  const float sg   = 1.0f / (1.0f + __expf(-5.0f * sraw)) + 1e-5f;
  const float sv   = exp2f(sg * 1.5849625007211562f) - 1.0f;  // 3^sg - 1
  const float coef = 0.3989422804014327f / sv;                // 1/(sqrt(2pi)*sv)
  const float n2   = 0.5f / (sv * sv);

  const int c0 = threadIdx.x * 8;
  float vals[8];
  float part = 0.f;
#pragma unroll
  for (int i = 0; i < 8; ++i) {
    float d = (float)(l - (c0 + i));
    float vv = coef * __expf(-(d * d) * n2);
    vals[i] = vv;
    part += vv;
  }
#pragma unroll
  for (int off = 16; off > 0; off >>= 1) part += __shfl_xor(part, off);

  __shared__ float red[8];
  if ((threadIdx.x & 31) == 0) red[threadIdx.x >> 5] = part;
  __syncthreads();
  float tot = 0.f;
#pragma unroll
  for (int w = 0; w < 8; ++w) tot += red[w];
  const float inv = 1.0f / (tot + 1e-8f);

  const size_t off0 = ((size_t)bh * LL + l) * LL + c0;
  v4f f0 = {vals[0] * inv, vals[1] * inv, vals[2] * inv, vals[3] * inv};
  v4f f1 = {vals[4] * inv, vals[5] * inv, vals[6] * inv, vals[7] * inv};
  __builtin_nontemporal_store(f0, (v4f*)(prior + off0));
  __builtin_nontemporal_store(f1, (v4f*)(prior + off0 + 4));
  v4f fs = {sv, sv, sv, sv};
  __builtin_nontemporal_store(fs, (v4f*)(sigb + off0));
  __builtin_nontemporal_store(fs, (v4f*)(sigb + off0 + 4));
}

extern "C" void kernel_launch(void* const* d_in, const int* in_sizes, int n_in,
                              void* d_out, int out_size, void* d_ws, size_t ws_size,
                              hipStream_t stream) {
  (void)in_sizes; (void)n_in; (void)out_size; (void)d_ws; (void)ws_size;
  const float* Q     = (const float*)d_in[0];
  const float* K     = (const float*)d_in[1];
  const float* V     = (const float*)d_in[2];
  const float* sigma = (const float*)d_in[3];

  float* outV   = (float*)d_out;
  float* series = outV + (size_t)BB * LL * HH * EE;            // 2,097,152
  float* prior  = series + (size_t)BB * HH * LL * LL;          // +67,108,864
  float* sigb   = prior + (size_t)BB * HH * LL * LL;           // +67,108,864

  anomaly_attn_kernel<<<dim3(BB * HH * 32), dim3(128), 0, stream>>>(Q, K, V, outV, series);
  anomaly_prior_kernel<<<dim3(BB * HH * LL), dim3(256), 0, stream>>>(sigma, prior, sigb);
}
